// MessagePassingLayer_38792144617921
// MI455X (gfx1250) — matrix operand order repaired and checked by reading, not compiled, not run on hardware
//
#include <hip/hip_runtime.h>

// ---------------------------------------------------------------------------
// MessagePassingLayer on MI455X (gfx1250), wave32, WMMA bf16 split-precision.
//
// - aggregate-then-transform (linearity of Linear+mean), degree-0 rows masked
//   to exact zero to match reference semantics.
// - edge gather/scatter: fp32 global atomics, 32 lanes/edge, fused counting.
// - node GEMMs: v_wmma_f32_16x16x32_bf16, bf16 hi/lo operand split
//   (Ah*Bh + Ah*Bl + Al*Bh, fp32 accumulate) ~ fp32 accuracy.
// - weights pre-swizzled to B-fragment-major layout -> 2x b128 loads/fragment.
// - X tile staged in LDS with padded stride (132) -> conflict-free ds_load_b128.
// ---------------------------------------------------------------------------

typedef __attribute__((ext_vector_type(16))) __bf16 v16bf;
typedef __attribute__((ext_vector_type(8)))  float  v8f;

#define DIMF 128
#define XPITCH 132   // LDS row pitch in floats (128 + 4 pad, bank-conflict-free)

// --- Weight pre-split + swizzle to WMMA B-fragment-major layout -------------
// Input: torch Linear W [out=128][in=K] row-major.
// Output: Bh/Bl indexed as [frag = kt*8+nt][lane 0..31][elem j 0..15], where
//   fragment (kt,nt) covers K-range [kt*32, kt*32+32) and N-range
//   [nt*16, nt*16+16); lane: lm=lane&15 (col), lh=lane>>4 (K half);
//   elem j -> B[k = kt*32 + lh*16 + j][n = nt*16 + lm] = W[n][k].
__global__ void split_weight_kernel(const float* __restrict__ W,
                                    __bf16* __restrict__ Bh,
                                    __bf16* __restrict__ Bl, int K) {
  int t = blockIdx.x * blockDim.x + threadIdx.x;
  int total = K * DIMF;
  if (t >= total) return;
  int j    = t & 15;
  int lane = (t >> 4) & 31;
  int f    = t >> 9;           // kt*8 + nt
  int nt   = f & 7;
  int kt   = f >> 3;
  int lm   = lane & 15;
  int lh   = lane >> 4;
  int k    = kt * 32 + lh * 16 + j;
  int n    = nt * 16 + lm;
  float w  = W[(size_t)n * K + k];
  __bf16 h = (__bf16)w;
  __bf16 l = (__bf16)(w - (float)h);
  Bh[t] = h;
  Bl[t] = l;
}

// --- Edge gather + scatter-add of 128-float rows, fused degree count --------
// 32 lanes per edge; each lane moves one float4 chunk (4 atomics).
__global__ void scatter_add_kernel(const float* __restrict__ feat,
                                   const int* __restrict__ src,
                                   const int* __restrict__ dst,
                                   float* __restrict__ agg,
                                   float* __restrict__ cnt, int E) {
  int t = blockIdx.x * blockDim.x + threadIdx.x;
  int e = t >> 5;
  int c = t & 31;
  if (e >= E) return;
  int d = dst[e];
  const float4 v = ((const float4*)(feat + (size_t)src[e] * DIMF))[c];
  float* a = agg + (size_t)d * DIMF + c * 4;
  atomicAdd(a + 0, v.x);
  atomicAdd(a + 1, v.y);
  atomicAdd(a + 2, v.z);
  atomicAdd(a + 3, v.w);
  if (c == 0) atomicAdd(&cnt[d], 1.0f);
}

// --- WMMA GEMM: Y[M x 128] = act( X @ W^T + b ) -----------------------------
// HALVES==1 (MSGMODE): X = agg/max(cnt,1), K=128, rows with cnt==0 -> 0.
// HALVES==2 (!MSGMODE): X = concat(X0, X1), K=256, relu epilogue.
// Block = 128 threads = 4 waves; block computes 64 rows x 128 cols.
template <int HALVES, bool MSGMODE>
__global__ __launch_bounds__(128) void gemm_wmma_kernel(
    const float* __restrict__ X0, const float* __restrict__ X1,
    const float* __restrict__ cnt,
    const __bf16* __restrict__ Wh, const __bf16* __restrict__ Wl,
    const float* __restrict__ bias, float* __restrict__ Y, int M) {
  __shared__ float Xs[64 * XPITCH];   // ~33 KB staging tile (fp32, padded)
  __shared__ float scnt[64];

  const int tid  = threadIdx.x;
  const int row0 = blockIdx.x * 64;
  const int wv   = tid >> 5;        // wave 0..3 -> rows [16*wv, 16*wv+15]
  const int lane = tid & 31;
  const int lm   = lane & 15;       // 0..15
  const int lh   = lane >> 4;       // 0 or 1

  const v16bf* __restrict__ WhF = (const v16bf*)Wh;  // fragment-major
  const v16bf* __restrict__ WlF = (const v16bf*)Wl;

  if (MSGMODE) {
    if (tid < 64) {
      int r = row0 + tid;
      scnt[tid] = (r < M) ? cnt[r] : 0.0f;
    }
  }

  v8f acc[8] = {};

  for (int half = 0; half < HALVES; ++half) {
    const float* Xsrc = (half == 0) ? X0 : X1;
    __syncthreads();  // scnt ready (iter 0) / previous half's reads done
    // Cooperative load: 64 rows x 32 float4 = 2048 float4, 16 per thread.
#pragma unroll
    for (int i = 0; i < 16; ++i) {
      int idx = tid + i * 128;
      int r   = idx >> 5;
      int c4  = idx & 31;
      int gr  = row0 + r;
      float4 v = make_float4(0.f, 0.f, 0.f, 0.f);
      if (gr < M) v = ((const float4*)(Xsrc + (size_t)gr * DIMF))[c4];
      if (MSGMODE) {
        float inv = 1.0f / fmaxf(scnt[r], 1.0f);
        v.x *= inv; v.y *= inv; v.z *= inv; v.w *= inv;
      }
      ((float4*)(Xs + r * XPITCH))[c4] = v;
    }
    __syncthreads();

#pragma unroll
    for (int kt = 0; kt < 4; ++kt) {
      // Per-lane A payload = 16 contiguous floats:
      //   Xs[row][kt*32 + lh*16 .. +16)
      // Vector elem order == k order (matches contiguous B fragment layout).
      const int rowl = wv * 16 + lm;
      const float* rp = Xs + rowl * XPITCH + kt * 32;
      float4 a0 = ((const float4*)rp)[lh * 4 + 0];
      float4 a1 = ((const float4*)rp)[lh * 4 + 1];
      float4 a2 = ((const float4*)rp)[lh * 4 + 2];
      float4 a3 = ((const float4*)rp)[lh * 4 + 3];
      float af[16] = {a0.x, a0.y, a0.z, a0.w, a1.x, a1.y, a1.z, a1.w,
                      a2.x, a2.y, a2.z, a2.w, a3.x, a3.y, a3.z, a3.w};
      v16bf ah, al;
#pragma unroll
      for (int j = 0; j < 16; ++j) {
        __bf16 h = (__bf16)af[j];
        ah[j] = h;
        al[j] = (__bf16)(af[j] - (float)h);
      }
      const int fbase = ((half * 4 + kt) * 8) * 32 + lane;
#pragma unroll
      for (int nt = 0; nt < 8; ++nt) {
        // One 32B fragment load per lane (2x global_load_b128), coalesced.
        v16bf bh = WhF[fbase + nt * 32];
        v16bf bl = WlF[fbase + nt * 32];
        // Split-precision: Ah*Bh + Ah*Bl + Al*Bh, fp32 accumulate.
        acc[nt] = __builtin_amdgcn_wmma_f32_16x16x32_bf16(
            false, ah, false, bh, (short)0, acc[nt], false, false);
        acc[nt] = __builtin_amdgcn_wmma_f32_16x16x32_bf16(
            false, ah, false, bl, (short)0, acc[nt], false, false);
        acc[nt] = __builtin_amdgcn_wmma_f32_16x16x32_bf16(
            false, al, false, bh, (short)0, acc[nt], false, false);
      }
    }
  }

  // Epilogue. D layout: VGPR r -> M = r + 8*lh, N = lm per 16x16 tile.
#pragma unroll
  for (int nt = 0; nt < 8; ++nt) {
    const int n  = nt * 16 + lm;
    const float bv = bias[n];
#pragma unroll
    for (int r = 0; r < 8; ++r) {
      int ml = wv * 16 + lh * 8 + r;
      int gr = row0 + ml;
      if (gr < M) {
        float y = acc[nt][r] + bv;
        if (MSGMODE)
          y = (scnt[ml] > 0.0f) ? y : 0.0f;  // degree-0 -> exact 0 (ref semantics)
        else
          y = fmaxf(y, 0.0f);                // relu
        Y[(size_t)gr * DIMF + n] = y;
      }
    }
  }
}

// ---------------------------------------------------------------------------
extern "C" void kernel_launch(void* const* d_in, const int* in_sizes, int n_in,
                              void* d_out, int out_size, void* d_ws,
                              size_t ws_size, hipStream_t stream) {
  const float* tile_h  = (const float*)d_in[0];
  const float* piece_h = (const float*)d_in[1];
  const int* tile_src  = (const int*)d_in[2];
  const int* piece_dst = (const int*)d_in[3];
  const int* piece_src = (const int*)d_in[4];
  const int* tile_dst  = (const int*)d_in[5];
  const int* t_src     = (const int*)d_in[6];
  const int* t_dst     = (const int*)d_in[7];
  const float* W_t2p = (const float*)d_in[8];  const float* b_t2p = (const float*)d_in[9];
  const float* W_pu  = (const float*)d_in[10]; const float* b_pu  = (const float*)d_in[11];
  const float* W_p2t = (const float*)d_in[12]; const float* b_p2t = (const float*)d_in[13];
  const float* W_tup = (const float*)d_in[14]; const float* b_tup = (const float*)d_in[15];
  const float* W_t2t = (const float*)d_in[16]; const float* b_t2t = (const float*)d_in[17];
  const float* W_tut = (const float*)d_in[18]; const float* b_tut = (const float*)d_in[19];

  const int NT = 50000, NP = 25000, EPT = 400000, ETT = 600000;

  float* tile_out  = (float*)d_out;                      // [NT,128]
  float* piece_out = (float*)d_out + (size_t)NT * DIMF;  // [NP,128]

  // Workspace carve-up.
  char* ws = (char*)d_ws;
  float* agg  = (float*)ws; ws += (size_t)NT * DIMF * sizeof(float);
  float* msg  = (float*)ws; ws += (size_t)NT * DIMF * sizeof(float);
  float* tmid = (float*)ws; ws += (size_t)NT * DIMF * sizeof(float);
  float* cntb = (float*)ws; ws += (size_t)NT * sizeof(float);
  __bf16* Wt2p_h = (__bf16*)ws; ws += 128 * DIMF * 2;
  __bf16* Wt2p_l = (__bf16*)ws; ws += 128 * DIMF * 2;
  __bf16* Wp2t_h = (__bf16*)ws; ws += 128 * DIMF * 2;
  __bf16* Wp2t_l = (__bf16*)ws; ws += 128 * DIMF * 2;
  __bf16* Wt2t_h = (__bf16*)ws; ws += 128 * DIMF * 2;
  __bf16* Wt2t_l = (__bf16*)ws; ws += 128 * DIMF * 2;
  __bf16* Wpu_h  = (__bf16*)ws; ws += 256 * DIMF * 2;
  __bf16* Wpu_l  = (__bf16*)ws; ws += 256 * DIMF * 2;
  __bf16* Wtup_h = (__bf16*)ws; ws += 256 * DIMF * 2;
  __bf16* Wtup_l = (__bf16*)ws; ws += 256 * DIMF * 2;
  __bf16* Wtut_h = (__bf16*)ws; ws += 256 * DIMF * 2;
  __bf16* Wtut_l = (__bf16*)ws; ws += 256 * DIMF * 2;

  // --- Weight hi/lo splits into fragment-major layout (cheap) ---
  {
    int tot128 = 128 * DIMF, tot256 = 256 * DIMF;
    split_weight_kernel<<<(tot128 + 255) / 256, 256, 0, stream>>>(W_t2p, Wt2p_h, Wt2p_l, 128);
    split_weight_kernel<<<(tot128 + 255) / 256, 256, 0, stream>>>(W_p2t, Wp2t_h, Wp2t_l, 128);
    split_weight_kernel<<<(tot128 + 255) / 256, 256, 0, stream>>>(W_t2t, Wt2t_h, Wt2t_l, 128);
    split_weight_kernel<<<(tot256 + 255) / 256, 256, 0, stream>>>(W_pu,  Wpu_h,  Wpu_l,  256);
    split_weight_kernel<<<(tot256 + 255) / 256, 256, 0, stream>>>(W_tup, Wtup_h, Wtup_l, 256);
    split_weight_kernel<<<(tot256 + 255) / 256, 256, 0, stream>>>(W_tut, Wtut_h, Wtut_l, 256);
  }

  // ---------------- Stage 1: tile -> piece ----------------
  hipMemsetAsync(agg,  0, (size_t)NP * DIMF * sizeof(float), stream);
  hipMemsetAsync(cntb, 0, (size_t)NP * sizeof(float), stream);
  scatter_add_kernel<<<(EPT * 32 + 255) / 256, 256, 0, stream>>>(
      tile_h, tile_src, piece_dst, agg, cntb, EPT);
  gemm_wmma_kernel<1, true><<<(NP + 63) / 64, 128, 0, stream>>>(
      agg, nullptr, cntb, Wt2p_h, Wt2p_l, b_t2p, msg, NP);
  gemm_wmma_kernel<2, false><<<(NP + 63) / 64, 128, 0, stream>>>(
      piece_h, msg, nullptr, Wpu_h, Wpu_l, b_pu, piece_out, NP);

  // ---------------- Stage 2: piece -> tile ----------------
  hipMemsetAsync(agg,  0, (size_t)NT * DIMF * sizeof(float), stream);
  hipMemsetAsync(cntb, 0, (size_t)NT * sizeof(float), stream);
  scatter_add_kernel<<<(EPT * 32 + 255) / 256, 256, 0, stream>>>(
      piece_out, piece_src, tile_dst, agg, cntb, EPT);
  gemm_wmma_kernel<1, true><<<(NT + 63) / 64, 128, 0, stream>>>(
      agg, nullptr, cntb, Wp2t_h, Wp2t_l, b_p2t, msg, NT);
  gemm_wmma_kernel<2, false><<<(NT + 63) / 64, 128, 0, stream>>>(
      tile_h, msg, nullptr, Wtup_h, Wtup_l, b_tup, tmid, NT);

  // ---------------- Stage 3: tile -> tile ----------------
  hipMemsetAsync(agg,  0, (size_t)NT * DIMF * sizeof(float), stream);
  hipMemsetAsync(cntb, 0, (size_t)NT * sizeof(float), stream);
  scatter_add_kernel<<<(ETT * 32 + 255) / 256, 256, 0, stream>>>(
      tmid, t_src, t_dst, agg, cntb, ETT);
  gemm_wmma_kernel<1, true><<<(NT + 63) / 64, 128, 0, stream>>>(
      agg, nullptr, cntb, Wt2t_h, Wt2t_l, b_t2t, msg, NT);
  gemm_wmma_kernel<2, false><<<(NT + 63) / 64, 128, 0, stream>>>(
      tmid, msg, nullptr, Wtut_h, Wtut_l, b_tut, tile_out, NT);
}
